// CrossAttention_90864328114869
// MI455X (gfx1250) — matrix operand order repaired and checked by reading, not compiled, not run on hardware
//
#include <hip/hip_runtime.h>

// ---------------------------------------------------------------------------
// CDNA5 (gfx1250) fused cross-attention: bf16 WMMA everywhere, f32 accumulate.
// Round 3: fix async-builtin pointer types (int4* with address spaces).
// ---------------------------------------------------------------------------

typedef __attribute__((ext_vector_type(16))) __bf16 v16bf;
typedef __attribute__((ext_vector_type(8)))  float  v8f;

union BFrag { v16bf v; unsigned u[8]; };
union CFrag { v8f   v; float    f[8]; };

#define HEADS 8
#define DHEAD 64
#define INNER 512
// scale * log2(e): softmax(sim*scale) computed in base-2 domain
#define SCALE_LOG2E (0.125f * 1.4426950408889634f)

static __device__ __forceinline__ unsigned short f2bf(float f) {
    __bf16 h = (__bf16)f;                       // RTNE fptrunc -> hw cvt
    return __builtin_bit_cast(unsigned short, h);
}

static __device__ __forceinline__ float fast_exp2(float x) {
#if __has_builtin(__builtin_amdgcn_exp2f)
    return __builtin_amdgcn_exp2f(x);
#else
    return exp2f(x);
#endif
}

// ---- async global->LDS copy (gfx1250 GLOBAL_LOAD_ASYNC_TO_LDS_B128) -------
#if __has_builtin(__builtin_amdgcn_global_load_async_to_lds_b128)
#define HAVE_ASYNC_LDS 1
typedef int v4i __attribute__((vector_size(16)));
typedef __attribute__((address_space(1))) v4i* gptr_v4i;   // global int4*
typedef __attribute__((address_space(3))) v4i* lptr_v4i;   // LDS int4*

static __device__ __forceinline__ void async_copy16(const void* gp, void* lp) {
    __builtin_amdgcn_global_load_async_to_lds_b128((gptr_v4i)gp, (lptr_v4i)lp, 0, 0);
}
static __device__ __forceinline__ void async_wait0() {
#if __has_builtin(__builtin_amdgcn_s_wait_asynccnt)
    __builtin_amdgcn_s_wait_asynccnt(0);
#else
    asm volatile("s_wait_asynccnt 0x0" ::: "memory");
#endif
}
#else
#define HAVE_ASYNC_LDS 0
#endif

// ---------------------------------------------------------------------------
// Tiled GEMM: C[M,N] = A[M,K] * B[K,N]; A is f32 or bf16, B (weights) is f32.
// OUT_MODE 0: bf16 row-major   OUT_MODE 1: bf16 scattered to [B,h,d,Nk] (V^T)
// OUT_MODE 2: f32 + bias (final projection)
// Block = 256 threads (8 waves, 4x2), tile BM=128 BN=128 BK=32.
// ---------------------------------------------------------------------------
template <int OUT_MODE, bool A_BF16>
__global__ __launch_bounds__(256) void gemm_wmma_bf16(
    const void* __restrict__ Ap, const float* __restrict__ Bw,
    const float* __restrict__ bias, void* __restrict__ Cp,
    int M, int N, int K)
{
    constexpr int BK = 32, LDT = BK + 2;          // even pad -> u32-aligned pairs
    __shared__ unsigned short lds_a[128 * LDT];   // [BM][BK]  (pairs along k)
    __shared__ unsigned short lds_b[128 * LDT];   // [BN][BK]  (B transposed)

    const int tid  = threadIdx.x;
    const int wave = tid >> 5, lane = tid & 31;
    const int half = lane >> 4, ln = lane & 15;
    const int wr = wave >> 1, wc = wave & 1;      // 4x2 wave grid
    const int row_base = blockIdx.y * 128;
    const int col_base = blockIdx.x * 128;

    CFrag acc[2][4];
#pragma unroll
    for (int tm = 0; tm < 2; ++tm)
#pragma unroll
        for (int tn = 0; tn < 4; ++tn)
#pragma unroll
            for (int v = 0; v < 8; ++v) acc[tm][tn].f[v] = 0.f;

    for (int k0 = 0; k0 < K; k0 += BK) {
        __syncthreads();
        // ---- stage A tile [128][32], converting to bf16 if needed ----
#pragma unroll
        for (int i = 0; i < 16; ++i) {
            int idx = tid + i * 256;
            int r = idx >> 5, c = idx & 31;
            unsigned short us;
            if (A_BF16)
                us = ((const unsigned short*)Ap)[(size_t)(row_base + r) * K + (k0 + c)];
            else
                us = f2bf(((const float*)Ap)[(size_t)(row_base + r) * K + (k0 + c)]);
            lds_a[r * LDT + c] = us;
        }
        // ---- stage B tile [32][128] transposed into [n][k] ----
#pragma unroll
        for (int i = 0; i < 16; ++i) {
            int idx = tid + i * 256;
            int kk = idx >> 7, n = idx & 127;
            lds_b[n * LDT + kk] = f2bf(Bw[(size_t)(k0 + kk) * N + (col_base + n)]);
        }
        __syncthreads();

        // ---- compute: 2 A-frags, 4 B-frags, 8 WMMAs ----
        BFrag af[2];
#pragma unroll
        for (int tm = 0; tm < 2; ++tm) {
            int m = wr * 32 + tm * 16 + ln;
#pragma unroll
            for (int v = 0; v < 8; ++v) {
                int kk = ((v < 4) ? 2 * v : 2 * v + 8) + 8 * half;   // A-frag k map
                af[tm].u[v] = *(const unsigned*)&lds_a[m * LDT + kk];
            }
        }
#pragma unroll
        for (int tn = 0; tn < 4; ++tn) {
            BFrag bf;
            int n = wc * 64 + tn * 16 + ln;
#pragma unroll
            for (int v = 0; v < 8; ++v) {
                int kk = ((v < 4) ? 2 * v : 2 * v + 8) + 8 * half;   // B-frag k map
                bf.u[v] = *(const unsigned*)&lds_b[n * LDT + kk];
            }
#pragma unroll
            for (int tm = 0; tm < 2; ++tm)
                acc[tm][tn].v = __builtin_amdgcn_wmma_f32_16x16x32_bf16(
                    false, af[tm].v, false, bf.v, (short)0, acc[tm][tn].v, false, false);
        }
    }

    // ---- epilogue ----
#pragma unroll
    for (int tm = 0; tm < 2; ++tm)
#pragma unroll
        for (int tn = 0; tn < 4; ++tn)
#pragma unroll
            for (int v = 0; v < 8; ++v) {
                int r = row_base + wr * 32 + tm * 16 + v + 8 * half;  // C-frag m map
                int c = col_base + wc * 64 + tn * 16 + ln;            // C-frag n map
                float val = acc[tm][tn].f[v];
                if constexpr (OUT_MODE == 0) {
                    ((unsigned short*)Cp)[(size_t)r * N + c] = f2bf(val);
                } else if constexpr (OUT_MODE == 1) {
                    // r = b*1024 + key, c = h*64 + d  ->  vt[((b*8+h)*64+d)*1024 + key]
                    int b = r >> 10, j = r & 1023;
                    int h = c >> 6,  d = c & 63;
                    ((unsigned short*)Cp)[((size_t)((b * 8 + h) * 64 + d)) * 1024 + j] = f2bf(val);
                } else {
                    ((float*)Cp)[(size_t)r * N + c] = val + bias[c];
                }
            }
}

// ---------------------------------------------------------------------------
// Fused flash attention per (b,h): Q[2048,64] bf16, K[1024,64] bf16,
// V^T[64,1024] bf16 -> O bf16 [B,Nq,512]. 8 waves x 16 Q rows, 64-key KV tiles.
// K/V^T tiles staged via GLOBAL_LOAD_ASYNC_TO_LDS_B128 when available.
// ---------------------------------------------------------------------------
__global__ __launch_bounds__(256) void attn_wmma_bf16(
    const unsigned short* __restrict__ qb, const unsigned short* __restrict__ kb,
    const unsigned short* __restrict__ vtb, unsigned short* __restrict__ ob,
    int Nq, int Nk)
{
    constexpr int LDK = 72;                       // 144 B row: 16B-aligned chunks
    constexpr int LDP = 66;                       // even pad
    __shared__ unsigned short k_lds[64 * LDK];    // [key][d]
    __shared__ unsigned short vt_lds[64 * LDK];   // [d][key]
    __shared__ unsigned short p_lds[8][16 * LDP]; // per-wave P re-layout

    const int bh = blockIdx.y;
    const int b = bh / HEADS, h = bh % HEADS;
    const int tid = threadIdx.x;
    const int wave = tid >> 5, lane = tid & 31;
    const int half = lane >> 4, ln = lane & 15;
    const int q_row0 = blockIdx.x * 128 + wave * 16;

    const unsigned short* qbase  = qb  + (size_t)b * Nq * INNER + h * DHEAD;
    const unsigned short* kbase  = kb  + (size_t)b * Nk * INNER + h * DHEAD;
    const unsigned short* vtbase = vtb + (size_t)bh * DHEAD * Nk;

    // Q A-frags live in registers for the whole kernel (2 frags = 16x64)
    BFrag qf[2];
#pragma unroll
    for (int ks = 0; ks < 2; ++ks)
#pragma unroll
        for (int v = 0; v < 8; ++v) {
            int kk = ks * 32 + ((v < 4) ? 2 * v : 2 * v + 8) + 8 * half;
            qf[ks].u[v] = *(const unsigned*)(qbase + (size_t)(q_row0 + ln) * INNER + kk);
        }

    CFrag o[4];
    float mrow[8], lrow[8];
#pragma unroll
    for (int tn = 0; tn < 4; ++tn)
#pragma unroll
        for (int v = 0; v < 8; ++v) o[tn].f[v] = 0.f;
#pragma unroll
    for (int v = 0; v < 8; ++v) { mrow[v] = -3.0e38f; lrow[v] = 0.f; }

    for (int j0 = 0; j0 < Nk; j0 += 64) {
        __syncthreads();
#if HAVE_ASYNC_LDS
        // ---- async stage: 512 x 16B chunks per tile, 2 per thread per tile --
#pragma unroll
        for (int i = 0; i < 2; ++i) {
            int idx = tid + i * 256;
            int r = idx >> 3, c = (idx & 7) * 8;     // 8 bf16 = 16 B
            async_copy16(kbase + (size_t)(j0 + r) * INNER + c, &k_lds[r * LDK + c]);
        }
#pragma unroll
        for (int i = 0; i < 2; ++i) {
            int idx = tid + i * 256;
            int d = idx >> 3, c = (idx & 7) * 8;
            async_copy16(vtbase + (size_t)d * Nk + j0 + c, &vt_lds[d * LDK + c]);
        }
        async_wait0();
#else
        // ---- fallback: synchronous u32 staging ----
#pragma unroll
        for (int i = 0; i < 8; ++i) {
            int idx = tid + i * 256;
            int r = idx >> 5, c2 = (idx & 31) * 2;
            *(unsigned*)&k_lds[r * LDK + c2] =
                *(const unsigned*)(kbase + (size_t)(j0 + r) * INNER + c2);
        }
#pragma unroll
        for (int i = 0; i < 8; ++i) {
            int idx = tid + i * 256;
            int d = idx >> 5, c2 = (idx & 31) * 2;
            *(unsigned*)&vt_lds[d * LDK + c2] =
                *(const unsigned*)(vtbase + (size_t)d * Nk + j0 + c2);
        }
#endif
        __syncthreads();

        // ---- S = Q * K^T  (4 n-tiles x 2 k-steps) ----
        CFrag s[4];
#pragma unroll
        for (int tn = 0; tn < 4; ++tn)
#pragma unroll
            for (int v = 0; v < 8; ++v) s[tn].f[v] = 0.f;
#pragma unroll
        for (int ks = 0; ks < 2; ++ks) {
#pragma unroll
            for (int tn = 0; tn < 4; ++tn) {
                BFrag bf;
                int key = tn * 16 + ln;
#pragma unroll
                for (int v = 0; v < 8; ++v) {
                    int kk = ks * 32 + ((v < 4) ? 2 * v : 2 * v + 8) + 8 * half;
                    bf.u[v] = *(const unsigned*)&k_lds[key * LDK + kk];
                }
                s[tn].v = __builtin_amdgcn_wmma_f32_16x16x32_bf16(
                    false, qf[ks].v, false, bf.v, (short)0, s[tn].v, false, false);
            }
        }

        // ---- online softmax update (base-2 domain) ----
#pragma unroll
        for (int tn = 0; tn < 4; ++tn)
#pragma unroll
            for (int v = 0; v < 8; ++v) s[tn].f[v] *= SCALE_LOG2E;

        float mnew[8], alpha[8];
#pragma unroll
        for (int v = 0; v < 8; ++v) {
            float x = fmaxf(fmaxf(s[0].f[v], s[1].f[v]), fmaxf(s[2].f[v], s[3].f[v]));
#pragma unroll
            for (int msk = 1; msk < 16; msk <<= 1) x = fmaxf(x, __shfl_xor(x, msk, 32));
            mnew[v]  = fmaxf(mrow[v], x);
            alpha[v] = fast_exp2(mrow[v] - mnew[v]);
            mrow[v]  = mnew[v];
        }
#pragma unroll
        for (int v = 0; v < 8; ++v) {
            float rs = 0.f;
            int m = v + 8 * half;
#pragma unroll
            for (int tn = 0; tn < 4; ++tn) {
                float p = fast_exp2(s[tn].f[v] - mnew[v]);
                rs += p;
                p_lds[wave][m * LDP + tn * 16 + ln] = f2bf(p);   // same-wave LDS
            }
#pragma unroll
            for (int msk = 1; msk < 16; msk <<= 1) rs += __shfl_xor(rs, msk, 32);
            lrow[v] = lrow[v] * alpha[v] + rs;
        }
#pragma unroll
        for (int tn = 0; tn < 4; ++tn)
#pragma unroll
            for (int v = 0; v < 8; ++v) o[tn].f[v] *= alpha[v];

        // ---- O += P * V  (A = P[16x64keys] from LDS, B = V^T tile) ----
#pragma unroll
        for (int ks = 0; ks < 2; ++ks) {
            BFrag af;
#pragma unroll
            for (int v = 0; v < 8; ++v) {
                int kk = ks * 32 + ((v < 4) ? 2 * v : 2 * v + 8) + 8 * half;
                af.u[v] = *(const unsigned*)&p_lds[wave][ln * LDP + kk];
            }
#pragma unroll
            for (int tn = 0; tn < 4; ++tn) {
                BFrag bf;
                int dcol = tn * 16 + ln;
#pragma unroll
                for (int v = 0; v < 8; ++v) {
                    int kk = ks * 32 + ((v < 4) ? 2 * v : 2 * v + 8) + 8 * half;
                    bf.u[v] = *(const unsigned*)&vt_lds[dcol * LDK + kk];
                }
                o[tn].v = __builtin_amdgcn_wmma_f32_16x16x32_bf16(
                    false, af.v, false, bf.v, (short)0, o[tn].v, false, false);
            }
        }
    }

    // ---- normalize and store bf16 [B,Nq,512] ----
#pragma unroll
    for (int tn = 0; tn < 4; ++tn)
#pragma unroll
        for (int v = 0; v < 8; ++v) {
            int m = v + 8 * half;
            int row = q_row0 + m;
            int col = h * DHEAD + tn * 16 + ln;
            ob[((size_t)b * Nq + row) * INNER + col] = f2bf(o[tn].f[v] / lrow[v]);
        }
}

// ---------------------------------------------------------------------------
extern "C" void kernel_launch(void* const* d_in, const int* in_sizes, int n_in,
                              void* d_out, int out_size, void* d_ws, size_t ws_size,
                              hipStream_t stream) {
    (void)in_sizes; (void)n_in; (void)out_size; (void)ws_size;
    const float* x   = (const float*)d_in[0];   // [4,2048,512]
    const float* ctx = (const float*)d_in[1];   // [4,1024,768]
    const float* Wq  = (const float*)d_in[2];   // [512,512]
    const float* Wk  = (const float*)d_in[3];   // [768,512]
    const float* Wv  = (const float*)d_in[4];   // [768,512]
    const float* Wo  = (const float*)d_in[5];   // [512,512]
    const float* bo  = (const float*)d_in[6];   // [512]

    const int B = 4, Nq = 2048, Dq = 512, Nk = 1024, Dc = 768;

    unsigned short* q_bf  = (unsigned short*)d_ws;                 // [B*Nq,512]
    unsigned short* k_bf  = q_bf  + (size_t)B * Nq * INNER;        // [B*Nk,512]
    unsigned short* vt_bf = k_bf  + (size_t)B * Nk * INNER;        // [B,h,64,Nk]
    unsigned short* ao_bf = vt_bf + (size_t)B * HEADS * DHEAD * Nk;// [B*Nq,512]

    dim3 blk(256);
    // 1) q = x @ Wq
    gemm_wmma_bf16<0, false><<<dim3(INNER / 128, (B * Nq) / 128), blk, 0, stream>>>(
        x, Wq, nullptr, q_bf, B * Nq, INNER, Dq);
    // 2) k = ctx @ Wk
    gemm_wmma_bf16<0, false><<<dim3(INNER / 128, (B * Nk) / 128), blk, 0, stream>>>(
        ctx, Wk, nullptr, k_bf, B * Nk, INNER, Dc);
    // 3) v = ctx @ Wv, stored transposed per head
    gemm_wmma_bf16<1, false><<<dim3(INNER / 128, (B * Nk) / 128), blk, 0, stream>>>(
        ctx, Wv, nullptr, vt_bf, B * Nk, INNER, Dc);
    // 4) fused flash attention
    attn_wmma_bf16<<<dim3(Nq / 128, B * HEADS), blk, 0, stream>>>(
        q_bf, k_bf, vt_bf, ao_bf, Nq, Nk);
    // 5) out = ao @ Wo + bo (f32)
    gemm_wmma_bf16<2, true><<<dim3(Dq / 128, (B * Nq) / 128), blk, 0, stream>>>(
        ao_bf, Wo, bo, (float*)d_out, B * Nq, Dq, INNER);
}
